// SequenceDistanceEmbeddingInitializer_79817672229229
// MI455X (gfx1250) — 8-bit matrix operand order repaired and checked by reading, compiled, not run on hardware
//
#include <hip/hip_runtime.h>
#include <stdint.h>

#define N_SEQ   2048
#define S_SITES 4096
#define K_DIM   (S_SITES * 4)
#define FP8_ONE 0x38u

#define BLK_M      128
#define BLK_N      128
#define KC         128
#define STEPS      (K_DIM / KC)
#define ROW_STRIDE (KC + 16)
#define BUF_BYTES  (2 * BLK_M * ROW_STRIDE)

#define NBLK       (N_SEQ / BLK_M)
#define NTRI       (NBLK * (NBLK + 1) / 2)

#define PACK_BYTES ((size_t)N_SEQ * K_DIM)
#define Z_OFF      PACK_BYTES
#define SQ_OFF     (Z_OFF  + (size_t)N_SEQ * 2 * sizeof(float))
#define MISC_OFF   (SQ_OFF + (size_t)N_SEQ * sizeof(float))

typedef int   v16i __attribute__((ext_vector_type(16)));
typedef float v8f  __attribute__((ext_vector_type(8)));

typedef int vi4_t __attribute__((vector_size(16)));
typedef __attribute__((address_space(1))) vi4_t* gvi4_p;
typedef __attribute__((address_space(3))) vi4_t* lvi4_p;

#if defined(__gfx1250__) && __has_builtin(__builtin_amdgcn_global_load_async_to_lds_b128)
#define ASYNC_LDS 1
#else
#define ASYNC_LDS 0
#endif

#if defined(__gfx1250__) && __has_builtin(__builtin_amdgcn_s_wait_asynccnt)
#define WAIT_ASYNC(n) __builtin_amdgcn_s_wait_asynccnt((short)(n))
#elif defined(__gfx1250__)
#define WAIT_ASYNC(n) asm volatile("s_wait_asynccnt %0" ::"n"(n) : "memory")
#else
#define WAIT_ASYNC(n)
#endif

__global__ void prep_kernel(const float* __restrict__ table,
                            const float* __restrict__ log_scale,
                            float2* __restrict__ z,
                            float*  __restrict__ sq,
                            float*  __restrict__ misc)
{
    int i = blockIdx.x * blockDim.x + threadIdx.x;
    if (i < N_SEQ) {
        float x = table[2 * i], y = table[2 * i + 1];
        float n   = sqrtf(x * x + y * y);
        float t   = tanhf(n);
        float inv = 1.0f / fmaxf(n, 1e-12f);
        float zx = x * inv * t, zy = y * inv * t;
        z[i]  = make_float2(zx, zy);
        sq[i] = zx * zx + zy * zy;
    }
    if (i == 0) {
        misc[0] = expf(log_scale[0]);
        misc[1] = 0.0f;
    }
}

__global__ void pack_kernel(const float4* __restrict__ in4,
                            uint32_t*     __restrict__ out)
{
    int i = blockIdx.x * blockDim.x + threadIdx.x;
    float4 f = in4[i];
    out[i] = (f.x > 0.5f ? FP8_ONE : 0u)
           | (f.y > 0.5f ? FP8_ONE : 0u) << 8
           | (f.z > 0.5f ? FP8_ONE : 0u) << 16
           | (f.w > 0.5f ? FP8_ONE : 0u) << 24;
}

__device__ __forceinline__ v16i load_a_frag(const uint8_t* p)
{
    uint2 q0 = *(const uint2*)(p);
    uint2 q1 = *(const uint2*)(p + 16);
    uint2 q2 = *(const uint2*)(p + 32);
    uint2 q3 = *(const uint2*)(p + 48);
    uint2 q4 = *(const uint2*)(p + 64);
    uint2 q5 = *(const uint2*)(p + 80);
    uint2 q6 = *(const uint2*)(p + 96);
    uint2 q7 = *(const uint2*)(p + 112);
    v16i a;
    a[0]=(int)q0.x;  a[1]=(int)q0.y;  a[2]=(int)q1.x;  a[3]=(int)q1.y;
    a[4]=(int)q2.x;  a[5]=(int)q2.y;  a[6]=(int)q3.x;  a[7]=(int)q3.y;
    a[8]=(int)q4.x;  a[9]=(int)q4.y;  a[10]=(int)q5.x; a[11]=(int)q5.y;
    a[12]=(int)q6.x; a[13]=(int)q6.y; a[14]=(int)q7.x; a[15]=(int)q7.y;
    return a;
}

__device__ __forceinline__ v16i load_b_frag(const uint8_t* p)
{
    uint2 q0 = *(const uint2*)(p);
    uint2 q1 = *(const uint2*)(p + 16);
    uint2 q2 = *(const uint2*)(p + 32);
    uint2 q3 = *(const uint2*)(p + 48);
    uint2 q4 = *(const uint2*)(p + 64);
    uint2 q5 = *(const uint2*)(p + 80);
    uint2 q6 = *(const uint2*)(p + 96);
    uint2 q7 = *(const uint2*)(p + 112);
    v16i b;
    b[0]=(int)q0.x;  b[1]=(int)q0.y;  b[2]=(int)q1.x;  b[3]=(int)q1.y;
    b[4]=(int)q2.x;  b[5]=(int)q2.y;  b[6]=(int)q3.x;  b[7]=(int)q3.y;
    b[8]=(int)q4.x;  b[9]=(int)q4.y;  b[10]=(int)q5.x; b[11]=(int)q5.y;
    b[12]=(int)q6.x; b[13]=(int)q6.y; b[14]=(int)q7.x; b[15]=(int)q7.y;
    return b;
}

__global__ void __launch_bounds__(256)
gram_mse_kernel(const uint8_t* __restrict__ pack,
                const float2*  __restrict__ z,
                const float*   __restrict__ sq,
                const float*   __restrict__ misc,
                float*         __restrict__ acc)
{
#if ASYNC_LDS
    __shared__ uint8_t lds[2 * BUF_BYTES];
#else
    __shared__ uint8_t lds[BUF_BYTES];
#endif

    const int tid  = threadIdx.x;
    const int lane = tid & 31;
    const int wave = tid >> 5;
    const int half = lane >> 4;
    const int l15  = lane & 15;

    int bi = 0, rem = blockIdx.x;
    while (rem >= NBLK - bi) { rem -= NBLK - bi; ++bi; }
    const int bj = bi + rem;
    const float weight = (bi == bj) ? 1.0f : 2.0f;

    const int blkRow = bi * BLK_M;
    const int blkCol = bj * BLK_N;

    const int rowIdx = (tid < BLK_M) ? (blkRow + tid) : (blkCol + tid - BLK_M);
    uint8_t* gsrc = const_cast<uint8_t*>(pack) + (size_t)rowIdx * K_DIM;
    const int lofs = tid * ROW_STRIDE;

    const int wr = wave >> 1;
    const int wc = wave & 1;
    const int aOfs0 = (wr * 32 + l15) * ROW_STRIDE + half * 8;
    const int aOfs1 = aOfs0 + 16 * ROW_STRIDE;
    const int bOfs0 = BLK_M * ROW_STRIDE + (wc * 64 + l15) * ROW_STRIDE + half * 8;

    v8f acc0[4] = {{0}, {0}, {0}, {0}};
    v8f acc1[4] = {{0}, {0}, {0}, {0}};

#if ASYNC_LDS
    #pragma unroll
    for (int i = 0; i < 9; ++i)
        __builtin_amdgcn_global_load_async_to_lds_b128(
            (gvi4_p)(gsrc + i * 16),
            (lvi4_p)(lds + lofs + i * 16), 0, 0);

    for (int step = 0; step < STEPS; ++step) {
        const int cur = step & 1;
        if (step + 1 < STEPS) {
            uint8_t* g = gsrc + (step + 1) * KC;
            uint8_t* l = lds + (cur ^ 1) * BUF_BYTES + lofs;
            #pragma unroll
            for (int i = 0; i < 9; ++i)
                __builtin_amdgcn_global_load_async_to_lds_b128(
                    (gvi4_p)(g + i * 16),
                    (lvi4_p)(l + i * 16), 0, 0);
            WAIT_ASYNC(9);
        } else {
            WAIT_ASYNC(0);
        }
        __syncthreads();

        const uint8_t* base = lds + cur * BUF_BYTES;
        v16i a0 = load_a_frag(base + aOfs0);
        v16i a1 = load_a_frag(base + aOfs1);
        #pragma unroll
        for (int ct = 0; ct < 4; ++ct) {
            v16i b = load_b_frag(base + bOfs0 + ct * 16 * ROW_STRIDE);
            acc0[ct] = __builtin_amdgcn_wmma_f32_16x16x128_fp8_fp8(
                           a0, b, (short)0, acc0[ct], false, false);
            acc1[ct] = __builtin_amdgcn_wmma_f32_16x16x128_fp8_fp8(
                           a1, b, (short)0, acc1[ct], false, false);
        }
        __syncthreads();
    }
#else
    uint4 st[9];
    #pragma unroll
    for (int i = 0; i < 9; ++i) st[i] = *(const uint4*)(gsrc + i * 16);

    for (int step = 0; step < STEPS; ++step) {
        #pragma unroll
        for (int i = 0; i < 9; ++i)
            *(uint4*)(lds + lofs + i * 16) = st[i];
        __syncthreads();
        if (step + 1 < STEPS) {
            const uint8_t* g = gsrc + (step + 1) * KC;
            #pragma unroll
            for (int i = 0; i < 9; ++i) st[i] = *(const uint4*)(g + i * 16);
        }
        v16i a0 = load_a_frag(lds + aOfs0);
        v16i a1 = load_a_frag(lds + aOfs1);
        #pragma unroll
        for (int ct = 0; ct < 4; ++ct) {
            v16i b = load_b_frag(lds + bOfs0 + ct * 16 * ROW_STRIDE);
            acc0[ct] = __builtin_amdgcn_wmma_f32_16x16x128_fp8_fp8(
                           a0, b, (short)0, acc0[ct], false, false);
            acc1[ct] = __builtin_amdgcn_wmma_f32_16x16x128_fp8_fp8(
                           a1, b, (short)0, acc1[ct], false, false);
        }
        __syncthreads();
    }
#endif

    const float scale = misc[0];
    const float invS  = 1.0f / (float)S_SITES;

    float2 zc4[4];
    float  omc4[4];
    #pragma unroll
    for (int ct = 0; ct < 4; ++ct) {
        int col = blkCol + wc * 64 + ct * 16 + l15;
        zc4[ct]  = z[col];
        omc4[ct] = 1.0f - sq[col];
    }

    float sum = 0.0f;
    #pragma unroll
    for (int rt = 0; rt < 2; ++rt) {
        #pragma unroll
        for (int v = 0; v < 8; ++v) {
            int row = blkRow + wr * 32 + rt * 16 + v + half * 8;
            float2 zr  = z[row];
            float  omr = 1.0f - sq[row];
            #pragma unroll
            for (int ct = 0; ct < 4; ++ct) {
                float m   = rt ? acc1[ct][v] : acc0[ct][v];
                float ham = ((float)S_SITES - m) * invS;
                float dx = zr.x - zc4[ct].x, dy = zr.y - zc4[ct].y;
                float arg = 1.0f + 2.0f * (dx * dx + dy * dy) / (omr * omc4[ct]);
                arg = fmaxf(arg, 1.0f + 1e-7f);
                float dist = logf(arg + sqrtf(arg * arg - 1.0f)) * scale;
                float e = ham - dist;
                sum += e * e;
            }
        }
    }

    sum *= weight;
    for (int off = 16; off > 0; off >>= 1)
        sum += __shfl_xor(sum, off, 32);
    if (lane == 0) atomicAdd(acc, sum);
}

__global__ void finalize_kernel(const float* __restrict__ misc,
                                float* __restrict__ out)
{
    out[0] = misc[1] * (1.0f / ((float)N_SEQ * (float)N_SEQ));
}

extern "C" void kernel_launch(void* const* d_in, const int* in_sizes, int n_in,
                              void* d_out, int out_size, void* d_ws, size_t ws_size,
                              hipStream_t stream)
{
    const float* data      = (const float*)d_in[0];
    const float* table     = (const float*)d_in[1];
    const float* log_scale = (const float*)d_in[2];

    uint8_t* ws   = (uint8_t*)d_ws;
    uint8_t* pack = ws;
    float2*  z    = (float2*)(ws + Z_OFF);
    float*   sq   = (float*)(ws + SQ_OFF);
    float*   misc = (float*)(ws + MISC_OFF);

    prep_kernel<<<(N_SEQ + 255) / 256, 256, 0, stream>>>(table, log_scale, z, sq, misc);

    const int packElems = (int)(PACK_BYTES / 4);
    pack_kernel<<<packElems / 256, 256, 0, stream>>>((const float4*)data,
                                                     (uint32_t*)pack);

    gram_mse_kernel<<<NTRI, 256, 0, stream>>>(pack, z, sq, misc, misc + 1);

    finalize_kernel<<<1, 1, 0, stream>>>(misc, (float*)d_out);
}
